// Attention_27410481283907
// MI455X (gfx1250) — matrix operand order repaired and checked by reading, not compiled, not run on hardware
//
#include <hip/hip_runtime.h>
#include <hip/hip_bf16.h>

typedef __bf16 bf16_t;
typedef bf16_t v16bf __attribute__((ext_vector_type(16)));
typedef bf16_t v8bf  __attribute__((ext_vector_type(8)));
typedef float  v8f   __attribute__((ext_vector_type(8)));

#define B_   16
#define C_   512
#define N_   1024
#define G_   8
#define CG_  64          // channels per group
#define O3_  1536        // 3*C
#define EPS_ 1e-5f
#define SCALE_ 0.044194173824159216f   // 1/sqrt(512)

// ---------------------------------------------------------------------------
// CDNA5 async LDS staging: GLOBAL_LOAD_ASYNC_TO_LDS_B128 (ASYNCcnt-tracked,
// CDNA5 ISA). Generic pointers to LDS carry the LDS
// byte address in their low 32 bits (aperture in the high bits), so
// (uint32_t)(uintptr_t)shared_ptr is a valid VDST LDS address.
// ---------------------------------------------------------------------------
__device__ __forceinline__ void stage_async_b128(void* lds_dst, const void* gsrc) {
    uint32_t lds_addr = (uint32_t)(uintptr_t)lds_dst;
    uint64_t gaddr    = (uint64_t)(uintptr_t)gsrc;
    asm volatile("global_load_async_to_lds_b128 %0, %1, off"
                 :: "v"(lds_addr), "v"(gaddr) : "memory");
}
__device__ __forceinline__ void async_wait_all() {
    asm volatile("s_wait_asynccnt 0x0" ::: "memory");
}

// ---------------------------------------------------------------------------
// WMMA fragment helpers (CDNA5 16x16x32 bf16 layouts, CDNA5 ISA)
// A (16x32, MxK): lane L holds row M=L%16; K = (L/16)*8 + {0..7} in v[0..7],
//                 +16 in v[8..15]  -> two contiguous 8-elem chunks, row-major.
// B (32x16, KxN): lane L holds col N=L%16; K = (L/16)*8 + {0..7} in v[0..7],
//                 +16 in v[8..15]  -> same K order as A.
// C/D (16x16 f32): lane 0-15: (M=r, N=lane); lane 16-31: (M=r+8, N=lane-16).
// ---------------------------------------------------------------------------
__device__ __forceinline__ v16bf load_a_frag(const bf16_t* row, int kbase, int lane) {
    int k0 = kbase + ((lane >> 4) << 3);
    v8bf lo = *(const v8bf*)(row + k0);
    v8bf hi = *(const v8bf*)(row + k0 + 16);
    return __builtin_shufflevector(lo, hi, 0,1,2,3,4,5,6,7,8,9,10,11,12,13,14,15);
}
__device__ __forceinline__ v16bf load_b_frag(const bf16_t* colrow, int kbase, int lane) {
    int k0 = kbase + ((lane >> 4) << 3);
    v8bf lo = *(const v8bf*)(colrow + k0);
    v8bf hi = *(const v8bf*)(colrow + k0 + 16);
    return __builtin_shufflevector(lo, hi, 0,1,2,3,4,5,6,7,8,9,10,11,12,13,14,15);
}
__device__ __forceinline__ v8f wmma_bf16(v16bf a, v16bf b, v8f c) {
    // (neg_a, A, neg_b, B, c_mod, C, reuse_a, reuse_b)
    return __builtin_amdgcn_wmma_f32_16x16x32_bf16(false, a, false, b, (short)0, c, false, false);
}

// ---------------------------------------------------------------------------
// Kernel 1: fp32 -> bf16 weight conversion
// ---------------------------------------------------------------------------
__global__ void cvt_w_kernel(const float* __restrict__ wq, const float* __restrict__ wp,
                             bf16_t* __restrict__ wq_bf, bf16_t* __restrict__ wp_bf) {
    int i = blockIdx.x * 256 + threadIdx.x;
    if (i < O3_ * C_) wq_bf[i] = (bf16_t)wq[i];
    if (i < C_ * C_)  wp_bf[i] = (bf16_t)wp[i];
}

// ---------------------------------------------------------------------------
// Kernel 2: GroupNorm, writes hn TRANSPOSED as bf16 [B, N, C]
// ---------------------------------------------------------------------------
__global__ void gn_kernel(const float* __restrict__ x, const float* __restrict__ gamma,
                          const float* __restrict__ beta, bf16_t* __restrict__ hnT) {
    __shared__ float rs[256], rs2[256];
    __shared__ float s_mean, s_rstd;
    int g = blockIdx.x, b = blockIdx.y, tid = threadIdx.x;
    const float* xg = x + ((size_t)b * C_ + (size_t)g * CG_) * N_;

    float sum = 0.f, sum2 = 0.f;
    for (int e = tid; e < CG_ * N_; e += 256) {
        float v = xg[e];
        sum += v; sum2 += v * v;
    }
    rs[tid] = sum; rs2[tid] = sum2;
    __syncthreads();
    for (int s = 128; s > 0; s >>= 1) {
        if (tid < s) { rs[tid] += rs[tid + s]; rs2[tid] += rs2[tid + s]; }
        __syncthreads();
    }
    if (tid == 0) {
        float m   = rs[0]  * (1.0f / (CG_ * N_));
        float var = rs2[0] * (1.0f / (CG_ * N_)) - m * m;
        s_mean = m;
        s_rstd = rsqrtf(var + EPS_);
    }
    __syncthreads();
    float mean = s_mean, rstd = s_rstd;
    for (int e = tid; e < CG_ * N_; e += 256) {
        int c = g * CG_ + (e >> 10);
        int n = e & (N_ - 1);
        float v = (xg[e] - mean) * rstd * gamma[c] + beta[c];
        hnT[((size_t)b * N_ + n) * C_ + c] = (bf16_t)v;
    }
}

// ---------------------------------------------------------------------------
// Kernel 3: QKV gemm  qkv[o,n] = sum_c wqkv[o,c]*hn[c,n] + b[o]
// Block tile 32(M) x 256(N); 8 waves, each wave a 32x32 register tile
// (2 A-frags x 2 B-frags -> 4 WMMA per K-step).  K = 512.
// scatters: o<C -> qT[B,N,C], o<2C -> kT[B,N,C], else -> v[B,C,N]
// ---------------------------------------------------------------------------
__global__ void qkv_kernel(const bf16_t* __restrict__ wq, const float* __restrict__ bqkv,
                           const bf16_t* __restrict__ hnT,
                           bf16_t* __restrict__ qT, bf16_t* __restrict__ kT,
                           bf16_t* __restrict__ vM) {
    __shared__ bf16_t sW[32 * C_];            // 32 KB
    int tid = threadIdx.x, lane = tid & 31, wave = tid >> 5;
    int b = blockIdx.z, o_base = blockIdx.y * 32, n0 = blockIdx.x * 256;

    { // async-stage 32x512 weight tile (row-major) into LDS
        const char* gsrc = (const char*)(wq + (size_t)o_base * C_);
        char* ldst = (char*)sW;
        for (int i = tid; i < 2048; i += 256)
            stage_async_b128(ldst + i * 16, gsrc + i * 16);
        async_wait_all();
    }
    __syncthreads();

    int n_base = n0 + wave * 32;
    const bf16_t* arow0 = sW + (size_t)(lane & 15) * C_;
    const bf16_t* arow1 = arow0 + (size_t)16 * C_;
    const bf16_t* brow0 = hnT + ((size_t)b * N_ + n_base + (lane & 15)) * C_;
    const bf16_t* brow1 = brow0 + (size_t)16 * C_;

    v8f acc[2][2] = {};
    for (int kc = 0; kc < C_; kc += 32) {
        __builtin_prefetch(brow0 + kc + 128, 0, 0);
        v16bf a0 = load_a_frag(arow0, kc, lane);
        v16bf a1 = load_a_frag(arow1, kc, lane);
        v16bf bm0 = load_b_frag(brow0, kc, lane);
        v16bf bm1 = load_b_frag(brow1, kc, lane);
        acc[0][0] = wmma_bf16(a0, bm0, acc[0][0]);
        acc[0][1] = wmma_bf16(a0, bm1, acc[0][1]);
        acc[1][0] = wmma_bf16(a1, bm0, acc[1][0]);
        acc[1][1] = wmma_bf16(a1, bm1, acc[1][1]);
    }

    int hl = lane >> 4;
    #pragma unroll
    for (int mi = 0; mi < 2; ++mi) {
        #pragma unroll
        for (int ni = 0; ni < 2; ++ni) {
            int ncol = n_base + ni * 16 + (lane & 15);
            #pragma unroll
            for (int r = 0; r < 8; ++r) {
                int o = o_base + mi * 16 + r + hl * 8;
                float val = acc[mi][ni][r] + bqkv[o];
                if (o < C_) {
                    qT[((size_t)b * N_ + ncol) * C_ + o] = (bf16_t)val;
                } else if (o < 2 * C_) {
                    kT[((size_t)b * N_ + ncol) * C_ + (o - C_)] = (bf16_t)val;
                } else {
                    vM[((size_t)b * C_ + (o - 2 * C_)) * N_ + ncol] = (bf16_t)val;
                }
            }
        }
    }
}

// ---------------------------------------------------------------------------
// Kernel 4: attention for one 16-query tile.
//   S(16x1024) = scale * Qt K   (8 score tiles/wave share one A-frag per K)
//   P = exp(S - rowmax)          (unnormalized, bf16 in LDS)
//   O^T(16x512) = (P V^T)/rowsum (4 out tiles/wave share one A-frag per K)
// dynamic LDS: 16KB sQ + 64KB sS + 32KB sP + reductions = 113.1 KB
// ---------------------------------------------------------------------------
__global__ void attn_kernel(const bf16_t* __restrict__ qT, const bf16_t* __restrict__ kT,
                            const bf16_t* __restrict__ vM, bf16_t* __restrict__ oT) {
    extern __shared__ char smem[];
    bf16_t* sQ     = (bf16_t*)smem;               // 16 x 512 bf16
    float*  sS     = (float*)(smem + 16384);      // 16 x 1024 f32
    bf16_t* sP     = (bf16_t*)(smem + 81920);     // 16 x 1024 bf16
    float*  red    = (float*)(smem + 114688);     // 16 x 16
    float*  rowmax = (float*)(smem + 115712);     // 16
    float*  rowsum = (float*)(smem + 115776);     // 16

    int tid = threadIdx.x, lane = tid & 31, wave = tid >> 5;
    int b = blockIdx.y, q0 = blockIdx.x * 16;
    int hl = lane >> 4;

    { // async-stage Q^T tile (16 rows x 512) into LDS
        const char* gsrc = (const char*)(qT + ((size_t)b * N_ + q0) * C_);
        for (int i = tid; i < 1024; i += 256)
            stage_async_b128(smem + i * 16, gsrc + i * 16);
        async_wait_all();
    }
    __syncthreads();

    // --- S = scale * Q^T K : wave w owns score column tiles [w*128, w*128+128)
    const bf16_t* arowQ = sQ + (size_t)(lane & 15) * C_;
    const bf16_t* browK = kT + ((size_t)b * N_ + wave * 128 + (lane & 15)) * C_;
    {
        v8f acc[8] = {};
        for (int kc = 0; kc < C_; kc += 32) {
            v16bf a = load_a_frag(arowQ, kc, lane);
            #pragma unroll
            for (int t = 0; t < 8; ++t) {
                v16bf bm = load_b_frag(browK + (size_t)t * 16 * C_, kc, lane);
                acc[t] = wmma_bf16(a, bm, acc[t]);
            }
        }
        #pragma unroll
        for (int t = 0; t < 8; ++t) {
            #pragma unroll
            for (int r = 0; r < 8; ++r) {
                int n = r + hl * 8;
                sS[n * N_ + (wave * 8 + t) * 16 + (lane & 15)] = acc[t][r] * SCALE_;
            }
        }
    }
    __syncthreads();

    // --- softmax over each of the 16 rows (16 threads per row)
    int rrow = tid >> 4, jj = tid & 15;
    float lmax = -1e30f;
    for (int m = jj; m < N_; m += 16) lmax = fmaxf(lmax, sS[rrow * N_ + m]);
    red[rrow * 16 + jj] = lmax;
    __syncthreads();
    if (jj == 0) {
        float mx = red[rrow * 16];
        for (int t = 1; t < 16; ++t) mx = fmaxf(mx, red[rrow * 16 + t]);
        rowmax[rrow] = mx;
    }
    __syncthreads();
    float mx = rowmax[rrow];
    float lsum = 0.f;
    for (int m = jj; m < N_; m += 16) {
        float e = __expf(sS[rrow * N_ + m] - mx);
        sP[rrow * N_ + m] = (bf16_t)e;
        lsum += e;
    }
    red[rrow * 16 + jj] = lsum;
    __syncthreads();
    if (jj == 0) {
        float s = 0.f;
        for (int t = 0; t < 16; ++t) s += red[rrow * 16 + t];
        rowsum[rrow] = s;
    }
    __syncthreads();

    // --- O^T = (P V^T)/rowsum : wave w owns channel tiles [w*64, w*64+64)
    const bf16_t* arowP = sP + (size_t)(lane & 15) * N_;
    const bf16_t* browV = vM + ((size_t)b * C_ + wave * 64 + (lane & 15)) * N_;
    {
        v8f acc[4] = {};
        for (int km = 0; km < N_; km += 32) {
            __builtin_prefetch(browV + km + 128, 0, 0);
            v16bf a = load_a_frag(arowP, km, lane);
            #pragma unroll
            for (int ct = 0; ct < 4; ++ct) {
                v16bf bm = load_b_frag(browV + (size_t)ct * 16 * N_, km, lane);
                acc[ct] = wmma_bf16(a, bm, acc[ct]);
            }
        }
        #pragma unroll
        for (int ct = 0; ct < 4; ++ct) {
            int c = wave * 64 + ct * 16 + (lane & 15);
            #pragma unroll
            for (int r = 0; r < 8; ++r) {
                int n = r + hl * 8;
                float val = acc[ct][r] / rowsum[n];
                oT[((size_t)b * N_ + q0 + n) * C_ + c] = (bf16_t)val;
            }
        }
    }
}

// ---------------------------------------------------------------------------
// Kernel 5: proj gemm + bias + residual
// Block tile 32(M) x 256(N); wave = 32x32 register tile.
//   out[b,o,n] = x[b,o,n] + bproj[o] + sum_c wproj[o,c]*O[c,n]
// ---------------------------------------------------------------------------
__global__ void proj_kernel(const bf16_t* __restrict__ wp, const float* __restrict__ bp,
                            const bf16_t* __restrict__ oT, const float* __restrict__ x,
                            float* __restrict__ out) {
    __shared__ bf16_t sW[32 * C_];            // 32 KB
    int tid = threadIdx.x, lane = tid & 31, wave = tid >> 5;
    int b = blockIdx.z, o_base = blockIdx.y * 32, n0 = blockIdx.x * 256;

    {
        const char* gsrc = (const char*)(wp + (size_t)o_base * C_);
        char* ldst = (char*)sW;
        for (int i = tid; i < 2048; i += 256)
            stage_async_b128(ldst + i * 16, gsrc + i * 16);
        async_wait_all();
    }
    __syncthreads();

    int n_base = n0 + wave * 32;
    const bf16_t* arow0 = sW + (size_t)(lane & 15) * C_;
    const bf16_t* arow1 = arow0 + (size_t)16 * C_;
    const bf16_t* brow0 = oT + ((size_t)b * N_ + n_base + (lane & 15)) * C_;
    const bf16_t* brow1 = brow0 + (size_t)16 * C_;

    v8f acc[2][2] = {};
    for (int kc = 0; kc < C_; kc += 32) {
        __builtin_prefetch(brow0 + kc + 128, 0, 0);
        v16bf a0 = load_a_frag(arow0, kc, lane);
        v16bf a1 = load_a_frag(arow1, kc, lane);
        v16bf bm0 = load_b_frag(brow0, kc, lane);
        v16bf bm1 = load_b_frag(brow1, kc, lane);
        acc[0][0] = wmma_bf16(a0, bm0, acc[0][0]);
        acc[0][1] = wmma_bf16(a0, bm1, acc[0][1]);
        acc[1][0] = wmma_bf16(a1, bm0, acc[1][0]);
        acc[1][1] = wmma_bf16(a1, bm1, acc[1][1]);
    }

    int hl = lane >> 4;
    #pragma unroll
    for (int mi = 0; mi < 2; ++mi) {
        #pragma unroll
        for (int ni = 0; ni < 2; ++ni) {
            int ncol = n_base + ni * 16 + (lane & 15);
            #pragma unroll
            for (int r = 0; r < 8; ++r) {
                int o = o_base + mi * 16 + r + hl * 8;
                size_t idx = ((size_t)b * C_ + o) * N_ + ncol;
                out[idx] = x[idx] + bp[o] + acc[mi][ni][r];
            }
        }
    }
}

// ---------------------------------------------------------------------------
extern "C" void kernel_launch(void* const* d_in, const int* in_sizes, int n_in,
                              void* d_out, int out_size, void* d_ws, size_t ws_size,
                              hipStream_t stream) {
    (void)in_sizes; (void)n_in; (void)out_size; (void)ws_size;
    const float* x      = (const float*)d_in[0];
    const float* gamma  = (const float*)d_in[1];
    const float* beta   = (const float*)d_in[2];
    const float* w_qkv  = (const float*)d_in[3];
    const float* b_qkv  = (const float*)d_in[4];
    const float* w_proj = (const float*)d_in[5];
    const float* b_proj = (const float*)d_in[6];
    float* out = (float*)d_out;

    char* ws = (char*)d_ws;
    size_t off = 0;
    bf16_t* wq_bf = (bf16_t*)(ws + off); off += (size_t)O3_ * C_ * 2;      // 1.5 MB
    bf16_t* wp_bf = (bf16_t*)(ws + off); off += (size_t)C_ * C_ * 2;       // 0.5 MB
    bf16_t* hnT   = (bf16_t*)(ws + off); off += (size_t)B_ * N_ * C_ * 2;  // 16 MB
    bf16_t* qT    = (bf16_t*)(ws + off); off += (size_t)B_ * N_ * C_ * 2;  // 16 MB
    bf16_t* kT    = (bf16_t*)(ws + off); off += (size_t)B_ * N_ * C_ * 2;  // 16 MB
    bf16_t* vM    = (bf16_t*)(ws + off); off += (size_t)B_ * C_ * N_ * 2;  // 16 MB
    bf16_t* oT    = (bf16_t*)(ws + off); off += (size_t)B_ * N_ * C_ * 2;  // 16 MB

    cvt_w_kernel<<<(O3_ * C_) / 256, 256, 0, stream>>>(w_qkv, w_proj, wq_bf, wp_bf);
    gn_kernel<<<dim3(G_, B_), 256, 0, stream>>>(x, gamma, beta, hnT);
    qkv_kernel<<<dim3(N_ / 256, O3_ / 32, B_), 256, 0, stream>>>(wq_bf, b_qkv, hnT, qT, kT, vM);

    (void)hipFuncSetAttribute((const void*)attn_kernel,
                              hipFuncAttributeMaxDynamicSharedMemorySize, 115840);
    attn_kernel<<<dim3(N_ / 16, B_), 256, 115840, stream>>>(qT, kT, vM, oT);

    proj_kernel<<<dim3(N_ / 256, C_ / 32, B_), 256, 0, stream>>>(wp_bf, b_proj, oT, x, out);
}
